// SimpleSSM_32375463477676
// MI455X (gfx1250) — matrix operand order repaired and checked by reading, not compiled, not run on hardware
//
#include <hip/hip_runtime.h>
#include <hip/hip_bf16.h>

// ---------------------------------------------------------------------------
// SimpleSSM for MI455X (gfx1250, wave32, WMMA).
//   GEMM1 (34.4 GF) and GEMM2 (17.2 GF) run on v_wmma_f32_16x16x32_bf16,
//   double-buffered LDS pipeline, 128x128 block tile, 8 WMMA/wave/k-step.
//   Middle stage (conv + silu + W_x proj + closed-form state mix) fused.
// ---------------------------------------------------------------------------

typedef __attribute__((ext_vector_type(16))) __bf16 v16bf;
typedef __attribute__((ext_vector_type(8)))  float  v8f;

#define H_DIM   1024
#define I_DIM   2048
#define N_ST    16
#define K_CONV  4
#define BATCH   2
#define SEQ     2048
#define ROWS    (BATCH * SEQ)        // 4096

// round-to-nearest-even f32 -> bf16 bit pack (scalar fallback)
__device__ __forceinline__ unsigned short f2bf(float f) {
    unsigned u = __float_as_uint(f);
    u += 0x7FFFu + ((u >> 16) & 1u);
    return (unsigned short)(u >> 16);
}

// pack two f32 -> one dword of 2x bf16; prefer v_cvt_pk_bf16_f32 when available
__device__ __forceinline__ unsigned pack2bf(float lo, float hi) {
#if __has_builtin(__builtin_amdgcn_cvt_pk_bf16_f32)
    auto p = __builtin_amdgcn_cvt_pk_bf16_f32(lo, hi);
    return __builtin_bit_cast(unsigned, p);
#else
    return (unsigned)f2bf(lo) | ((unsigned)f2bf(hi) << 16);
#endif
}

__device__ __forceinline__ uint2 pack4bf(float4 v) {
    return make_uint2(pack2bf(v.x, v.y), pack2bf(v.z, v.w));
}

// ---------------------------------------------------------------------------
// C[M,N] = A[M,K] * B[N,K]^T   (A: f32 or bf16 row-major, B: f32 row-major,
// C: f32 row-major). bf16 WMMA, f32 accumulate.
// Block tile 128x128, BK=32, double-buffered LDS. 8 waves arranged
// 4(M) x 2(N); each wave owns 32x64 = 2x4 WMMA accumulators.
// ---------------------------------------------------------------------------
template <bool A_BF16>
__global__ __launch_bounds__(256)
void gemm_bf16_wmma(const void* __restrict__ Ap, const float* __restrict__ Bp,
                    float* __restrict__ Cp, int M, int N, int K)
{
    constexpr int BM = 128, BN = 128, BK = 32;
    constexpr int LDSW = 48;                       // halves per row (96B, 16B-aligned)
    __shared__ __align__(16) unsigned short As[2][BM * LDSW];
    __shared__ __align__(16) unsigned short Bs[2][BN * LDSW];

    const int tid   = threadIdx.x;
    const int wave  = tid >> 5;
    const int lane  = tid & 31;
    const int waveM = wave >> 1;                   // 4 waves along M (32 rows)
    const int waveN = wave & 1;                    // 2 waves along N (64 cols)
    const int tileM = blockIdx.y * BM;
    const int tileN = blockIdx.x * BN;
    const int hs    = lane >> 4;                   // lane half-select
    const int l16   = lane & 15;

    v8f acc[2][4];
    #pragma unroll
    for (int am = 0; am < 2; ++am)
        #pragma unroll
        for (int bn = 0; bn < 4; ++bn)
            #pragma unroll
            for (int r = 0; r < 8; ++r) acc[am][bn][r] = 0.f;

    // staging registers (global -> reg -> LDS pipeline)
    float4 aS32[4];
    uint4  aS16[2];
    float4 bS[4];

    auto load_tiles = [&](int k0) {
        if constexpr (A_BF16) {
            const unsigned short* A16 = (const unsigned short*)Ap;
            #pragma unroll
            for (int i = 0; i < 2; ++i) {          // 512 chunks of 8 halves
                int idx = tid + i * 256;
                int r = idx >> 2, c8 = (idx & 3) * 8;
                aS16[i] = *(const uint4*)(A16 + (size_t)(tileM + r) * K + k0 + c8);
            }
        } else {
            const float* A32 = (const float*)Ap;
            #pragma unroll
            for (int i = 0; i < 4; ++i) {          // 1024 float4 chunks
                int idx = tid + i * 256;
                int r = idx >> 3, c4 = (idx & 7) * 4;
                aS32[i] = *(const float4*)(A32 + (size_t)(tileM + r) * K + k0 + c4);
            }
        }
        #pragma unroll
        for (int i = 0; i < 4; ++i) {              // 1024 float4 chunks
            int idx = tid + i * 256;
            int r = idx >> 3, c4 = (idx & 7) * 4;
            bS[i] = *(const float4*)(Bp + (size_t)(tileN + r) * K + k0 + c4);
        }
    };

    auto store_tiles = [&](int buf) {
        if constexpr (A_BF16) {
            #pragma unroll
            for (int i = 0; i < 2; ++i) {
                int idx = tid + i * 256;
                int r = idx >> 2, c8 = (idx & 3) * 8;
                *(uint4*)&As[buf][r * LDSW + c8] = aS16[i];
            }
        } else {
            #pragma unroll
            for (int i = 0; i < 4; ++i) {
                int idx = tid + i * 256;
                int r = idx >> 3, c4 = (idx & 7) * 4;
                *(uint2*)&As[buf][r * LDSW + c4] = pack4bf(aS32[i]);
            }
        }
        #pragma unroll
        for (int i = 0; i < 4; ++i) {
            int idx = tid + i * 256;
            int r = idx >> 3, c4 = (idx & 7) * 4;
            *(uint2*)&Bs[buf][r * LDSW + c4] = pack4bf(bS[i]);
        }
    };

    // prologue: stage tile 0
    load_tiles(0);
    store_tiles(0);
    __syncthreads();

    const int nk = K / BK;
    for (int it = 0; it < nk; ++it) {
        const int cur = it & 1;
        const int nxt = cur ^ 1;

        // issue next tile's global loads early (latency hidden by WMMA)
        if (it + 1 < nk) load_tiles((it + 1) * BK);

        // ---- fragments per ISA 16-bit layouts, from buffer `cur` ----
        union Frag { v16bf v; uint4 q[2]; } a[2], b[4];
        #pragma unroll
        for (int am = 0; am < 2; ++am) {
            // A 16x32: lane<16 -> M=l16, K {0..7,16..23}; lane>=16 -> K {8..15,24..31}
            int row = waveM * 32 + am * 16 + l16;
            a[am].q[0] = *(const uint4*)&As[cur][row * LDSW + hs * 8];
            a[am].q[1] = *(const uint4*)&As[cur][row * LDSW + 16 + hs * 8];
        }
        #pragma unroll
        for (int bn = 0; bn < 4; ++bn) {
            // B 32x16: lane<16 -> N=l16, K {0..7,16..23}; lane>=16 -> K {8..15,24..31}
            int col = waveN * 64 + bn * 16 + l16;
            b[bn].q[0] = *(const uint4*)&Bs[cur][col * LDSW + hs * 8];
            b[bn].q[1] = *(const uint4*)&Bs[cur][col * LDSW + 16 + hs * 8];
        }
        #pragma unroll
        for (int am = 0; am < 2; ++am)
            #pragma unroll
            for (int bn = 0; bn < 4; ++bn)
                acc[am][bn] = __builtin_amdgcn_wmma_f32_16x16x32_bf16(
                    false, a[am].v, false, b[bn].v,
                    (short)0, acc[am][bn], false, false);

        if (it + 1 < nk) {
            store_tiles(nxt);       // other buffer: no race with cur readers
            __syncthreads();
        }
    }

    // ---- store C (16x16 f32 layout: VGPR r -> M = r, upper lanes +8) ----
    #pragma unroll
    for (int am = 0; am < 2; ++am)
        #pragma unroll
        for (int bn = 0; bn < 4; ++bn) {
            int m0 = tileM + waveM * 32 + am * 16 + hs * 8;
            int n  = tileN + waveN * 64 + bn * 16 + l16;
            float* cp = Cp + (size_t)m0 * N + n;
            #pragma unroll
            for (int r = 0; r < 8; ++r)
                cp[r * N] = acc[am][bn][r];
        }
}

// ---------------------------------------------------------------------------
// Fused middle stage. One block per (b,s) row; 256 threads x 8 channels.
//   xc = silu(depthwise_causal_conv4(xp));  sz = silu(z)
//   xproj = W_x . xc  (33 outputs via wave shuffle reduce + LDS f32 atomics)
//   y_i = xc_i * (sum_n B_n C_n exp(delta * -exp(A_log[i,n])) + D_i)
//   emit (y * sz) as bf16 for GEMM2.
// ---------------------------------------------------------------------------
__global__ __launch_bounds__(256)
void ssm_mid(const float* __restrict__ xz, const float* __restrict__ cw,
             const float* __restrict__ Wx, const float* __restrict__ Alog,
             const float* __restrict__ Dv, unsigned short* __restrict__ yz)
{
    __shared__ float red[33];
    __shared__ float bc[N_ST];                     // B[n]*C[n] precomputed
    const int tid  = threadIdx.x;
    const int lane = tid & 31;
    const int row  = blockIdx.x;
    const int b    = row / SEQ;
    const int s    = row % SEQ;

    if (tid < 33) red[tid] = 0.f;
    __syncthreads();

    const float* xzrow = xz + (size_t)row * (2 * I_DIM);

    float xc[8], sz[8];
    #pragma unroll
    for (int t = 0; t < 8; ++t) {
        int c = tid + 256 * t;
        float a = 0.f;
        #pragma unroll
        for (int k = 0; k < K_CONV; ++k) {         // causal conv, left pad 3
            int sk = s + k - (K_CONV - 1);
            if (sk >= 0)
                a += xz[(size_t)(b * SEQ + sk) * (2 * I_DIM) + c] * cw[c * K_CONV + k];
        }
        xc[t] = a / (1.f + __expf(-a));            // silu
        float zv = xzrow[I_DIM + c];
        sz[t] = zv / (1.f + __expf(-zv));
    }

    // 33 projections of xc against W_x rows
    for (int p = 0; p < 33; ++p) {
        const float* wr = Wx + (size_t)p * I_DIM;
        float pv = 0.f;
        #pragma unroll
        for (int t = 0; t < 8; ++t) pv += xc[t] * wr[tid + 256 * t];
        #pragma unroll
        for (int off = 16; off > 0; off >>= 1) pv += __shfl_xor(pv, off, 32);
        if (lane == 0) atomicAdd(&red[p], pv);     // ds_add_f32
    }
    __syncthreads();

    if (tid < N_ST) bc[tid] = red[1 + tid] * red[17 + tid];
    __syncthreads();

    float d0    = red[0];
    float delta = (d0 > 20.f) ? d0 : log1pf(__expf(d0));   // softplus

    #pragma unroll
    for (int t = 0; t < 8; ++t) {
        int c = tid + 256 * t;
        float ssum = 0.f;
        #pragma unroll
        for (int n = 0; n < N_ST; ++n) {
            float An = -__expf(Alog[(size_t)c * N_ST + n]);
            ssum += bc[n] * __expf(delta * An);
        }
        float y = xc[t] * (ssum + Dv[c]);
        yz[(size_t)row * I_DIM + c] = f2bf(y * sz[t]);
    }
}

// ---------------------------------------------------------------------------
extern "C" void kernel_launch(void* const* d_in, const int* in_sizes, int n_in,
                              void* d_out, int out_size, void* d_ws, size_t ws_size,
                              hipStream_t stream)
{
    (void)in_sizes; (void)n_in; (void)out_size; (void)ws_size;
    const float* x    = (const float*)d_in[0];   // (B,S,H)
    const float* Win  = (const float*)d_in[1];   // (2I,H)
    const float* cw   = (const float*)d_in[2];   // (I,1,K)
    const float* Wx   = (const float*)d_in[3];   // (33,I)
    const float* Alog = (const float*)d_in[4];   // (I,N)
    const float* Dv   = (const float*)d_in[5];   // (I)
    const float* Wout = (const float*)d_in[6];   // (H,I)
    float* out = (float*)d_out;                  // (B,S,H) f32

    float*          xzbuf = (float*)d_ws;                                  // 4096*4096 f32 (64MB)
    unsigned short* yzbuf = (unsigned short*)(xzbuf + (size_t)ROWS * 2 * I_DIM); // 4096*2048 bf16

    dim3 blk(256);

    // GEMM1: xz[4096,4096] = x[4096,1024] . Win[4096,1024]^T
    gemm_bf16_wmma<false><<<dim3((2 * I_DIM) / 128, ROWS / 128), blk, 0, stream>>>(
        (const void*)x, Win, xzbuf, ROWS, 2 * I_DIM, H_DIM);

    // conv + silu + projection + state mix -> yz (bf16)
    ssm_mid<<<dim3(ROWS), blk, 0, stream>>>(xzbuf, cw, Wx, Alog, Dv, yzbuf);

    // GEMM2: out[4096,1024] = yz[4096,2048](bf16) . Wout[1024,2048]^T
    gemm_bf16_wmma<true><<<dim3(H_DIM / 128, ROWS / 128), blk, 0, stream>>>(
        (const void*)yzbuf, Wout, out, ROWS, H_DIM, I_DIM);
}
